// Model_73684458930805
// MI455X (gfx1250) — matrix operand order repaired and checked by reading, not compiled, not run on hardware
//
#include <hip/hip_runtime.h>
#include <hip/hip_bf16.h>

typedef __attribute__((ext_vector_type(16))) _Float16 v16h;
typedef __attribute__((ext_vector_type(8)))  float    v8f;

#define BATCH   65536
#define LEAD    24
#define MEMBERS 51
#define NPRED   6
#define HID     128
#define NOUT    960
#define INV_SQRT2PI 0.3989422804014327f

// -------- workspace layout (in _Float16 units) ---------------------------
// Weights stored PRE-FRAGMENTED in WMMA B-layout:
//   frag(nt,kt) : 32 lanes x 16 contiguous halves, index ((nt*KT+kt)*32+lane)*16
#define OFF_WIN  0                          // 16 tiles  (KT=2, NT=8)   = 8192
#define OFF_WS   (8192)                     // 4 layers x 32 tiles      = 65536
#define OFF_WOUT (8192 + 65536)             // 240 tiles (KT=4, NT=60)  = 122880
#define OFF_A    (OFF_WOUT + 122880)        // BATCH x 64  (feat48|p6|pad)
#define OFF_H    (OFF_A + (long)BATCH * 64) // BATCH x 128 hidden (f16)

// =========================================================================
// Fast math helpers
// =========================================================================
__device__ __forceinline__ float frcp(float x)            { return __builtin_amdgcn_rcpf(x); }
__device__ __forceinline__ float fdivf(float a, float b)  { return a * __builtin_amdgcn_rcpf(b); }
__device__ __forceinline__ float silu(float x)            { return x * frcp(1.f + __expf(-x)); }
__device__ __forceinline__ float softplusf(float x) {
    return x > 20.f ? x : __logf(1.f + __expf(x));
}
__device__ __forceinline__ float gsel5(const float* a, int k) {
    float r = a[0];
#pragma unroll
    for (int i = 1; i < 5; ++i) r = (k == i) ? a[i] : r;
    return r;
}

// =========================================================================
// CDNA5 async global->LDS staging (ASYNCcnt path)
// =========================================================================
__device__ __forceinline__ void async_load_b128(unsigned lds_off, unsigned gvoff,
                                                unsigned long long sbase) {
    asm volatile("global_load_async_to_lds_b128 %0, %1, %2"
                 :: "v"(lds_off), "v"(gvoff), "s"(sbase) : "memory");
}
__device__ __forceinline__ void async_load_b64(unsigned lds_off, unsigned gvoff,
                                               unsigned long long sbase) {
    asm volatile("global_load_async_to_lds_b64 %0, %1, %2"
                 :: "v"(lds_off), "v"(gvoff), "s"(sbase) : "memory");
}
__device__ __forceinline__ void wait_async0() {
    asm volatile("s_wait_asynccnt 0" ::: "memory");
}
__device__ __forceinline__ unsigned lds_offset(const void* p) {
    return (unsigned)(unsigned long long)p;   // LDS aperture: low 32 bits = offset
}

// =========================================================================
// WMMA helpers (per CDNA5 ISA 7.12.2 layouts, wave32)
// =========================================================================
__device__ __forceinline__ v8f wmma_f16(v16h a, v16h b, v8f c) {
    return __builtin_amdgcn_wmma_f32_16x16x32_f16(
        false, a, false, b, (short)0, c, false, false);
}

// A fragment from LDS: 16x32 f16; lane L: M=L&15, ISA K-striping.
__device__ __forceinline__ v16h load_a_frag_lds(const _Float16* S, int ld,
                                                int k0, int lane) {
    int M  = lane & 15;
    int hi = lane >> 4;
    v16h a;
#pragma unroll
    for (int j = 0; j < 16; ++j) {
        int K = ((j >> 3) << 4) + (j & 7) + 8 * hi;
        a[j] = S[M * ld + k0 + K];
    }
    return a;
}

// B fragment from pre-fragmented weights: one 32B contiguous load per lane.
__device__ __forceinline__ v16h load_b_frag_pre(const _Float16* Wf, int KT,
                                                int kt, int nt, int lane) {
    return *(const v16h*)(Wf + (((nt * KT) + kt) * 32 + lane) * 16);
}

// =========================================================================
// Kernel 1: one-time weight repack f32 -> f16 fragment layout.
// One thread per (tile, lane): 384 tiles x 32 lanes = 12288 threads.
// =========================================================================
__global__ void prep_weights(const float* __restrict__ Win,
                             const float* __restrict__ Ws0, const float* __restrict__ Ws1,
                             const float* __restrict__ Ws2, const float* __restrict__ Ws3,
                             const float* __restrict__ Wout,
                             _Float16* __restrict__ wsh) {
    int idx  = blockIdx.x * blockDim.x + threadIdx.x;
    int lane = idx & 31;
    int tile = idx >> 5;                  // 0..383
    int hi   = lane >> 4;
    int N    = lane & 15;

    const float* W; _Float16* dst; int ldn, kbase, ncol, krows;
    if (tile < 16) {                      // W_in: KT=2, NT=8, rows 54 (pad->64)
        int nt = tile >> 1, kt = tile & 1;
        W = Win; ldn = 128; kbase = kt * 32; ncol = nt * 16 + N; krows = 54;
        dst = wsh + OFF_WIN + (long)tile * 512 + lane * 16;
    } else if (tile < 144) {              // Ws[4]: KT=4, NT=8 each
        int t2 = tile - 16;
        int layer = t2 >> 5, within = t2 & 31;
        int nt = within >> 2, kt = within & 3;
        W = (layer == 0) ? Ws0 : (layer == 1) ? Ws1 : (layer == 2) ? Ws2 : Ws3;
        ldn = 128; kbase = kt * 32; ncol = nt * 16 + N; krows = 128;
        dst = wsh + OFF_WS + (long)t2 * 512 + lane * 16;
    } else {                              // W_out: KT=4, NT=60
        int t3 = tile - 144;
        int nt = t3 >> 2, kt = t3 & 3;
        W = Wout; ldn = 960; kbase = kt * 32; ncol = nt * 16 + N; krows = 128;
        dst = wsh + OFF_WOUT + (long)t3 * 512 + lane * 16;
    }
    v16h v;
#pragma unroll
    for (int j = 0; j < 16; ++j) {
        int K = kbase + ((j >> 3) << 4) + (j & 7) + 8 * hi;
        v[j] = (_Float16)((K < krows) ? W[K * ldn + ncol] : 0.f);
    }
    *(v16h*)dst = v;
}

// =========================================================================
// Kernel 2: mean/std over 51 members — one wave32 per (b,lead) row
// =========================================================================
__global__ void stats_kernel(const float* __restrict__ x,
                             _Float16* __restrict__ A16) {
    int wid  = threadIdx.x >> 5;
    int lane = threadIdx.x & 31;
    long row = (long)blockIdx.x * 8 + wid;
    const float* xr = x + row * MEMBERS;
    float v0 = xr[lane];
    float v1 = (lane < MEMBERS - 32) ? xr[lane + 32] : 0.f;
    float s  = v0 + v1;
    float sq = v0 * v0 + v1 * v1;
#pragma unroll
    for (int off = 16; off >= 1; off >>= 1) {
        s  += __shfl_xor(s,  off, 32);
        sq += __shfl_xor(sq, off, 32);
    }
    if (lane == 0) {
        float mean = s * (1.f / 51.f);
        float var  = (sq - 51.f * mean * mean) * (1.f / 50.f);   // ddof=1
        float sd   = sqrtf(fmaxf(var, 0.f));
        int b = (int)(row / LEAD);
        int l = (int)(row % LEAD);
        A16[(long)b * 64 + 2 * l]     = (_Float16)mean;
        A16[(long)b * 64 + 2 * l + 1] = (_Float16)sd;
    }
}

// =========================================================================
// Kernel 3: append predictors + zero pad into A[B,64] cols 48..63
// =========================================================================
__global__ void pack_p(const float* __restrict__ p, _Float16* __restrict__ A16) {
    int idx = blockIdx.x * blockDim.x + threadIdx.x;   // BATCH*16
    int b = idx >> 4;
    int c = idx & 15;
    float v = (c < NPRED) ? p[b * NPRED + c] : 0.f;
    A16[(long)b * 64 + 48 + c] = (_Float16)v;
}

// =========================================================================
// Kernel 4: input layer + 4 SiLU skip blocks.  16 rows per 256-thread block.
// =========================================================================
__global__ void __launch_bounds__(256) mlp_kernel(
        const _Float16* __restrict__ wsh,
        const float* __restrict__ b_in,
        const float* __restrict__ bs0, const float* __restrict__ bs1,
        const float* __restrict__ bs2, const float* __restrict__ bs3,
        _Float16* __restrict__ H16,
        const _Float16* __restrict__ A16) {
    __shared__ _Float16 lds_a[16 * 64];
    __shared__ _Float16 lds_h[16 * 128];

    int tid  = threadIdx.x;
    int lane = tid & 31;
    int w    = tid >> 5;
    int r0   = blockIdx.x * 16;
    int N    = lane & 15;
    int hi   = lane >> 4;
    int n0   = w * 16;

    // async stage A tile [16 x 64] f16 (2 KB): 8 bytes per thread
    async_load_b64(lds_offset(lds_a) + tid * 8, (unsigned)(tid * 8),
                   (unsigned long long)(A16 + (long)r0 * 64));
    wait_async0();
    __syncthreads();

    // ---------------- input layer: h = silu(A @ Win + b_in) -------------
    v8f acc;
    {
        float bv = b_in[n0 + N];
#pragma unroll
        for (int r = 0; r < 8; ++r) acc[r] = bv;
    }
#pragma unroll
    for (int kt = 0; kt < 2; ++kt) {
        v16h a = load_a_frag_lds(lds_a, 64, kt * 32, lane);
        v16h b = load_b_frag_pre(wsh + OFF_WIN, 2, kt, w, lane);
        acc = wmma_f16(a, b, acc);
    }
#pragma unroll
    for (int r = 0; r < 8; ++r)
        lds_h[(r + 8 * hi) * 128 + n0 + N] = (_Float16)silu(acc[r]);

    // ---------------- 4 skip blocks: h = silu(h @ Ws + bs) + h ----------
#pragma unroll
    for (int layer = 0; layer < 4; ++layer) {
        __syncthreads();
        v16h a0 = load_a_frag_lds(lds_h, 128,  0, lane);
        v16h a1 = load_a_frag_lds(lds_h, 128, 32, lane);
        v16h a2 = load_a_frag_lds(lds_h, 128, 64, lane);
        v16h a3 = load_a_frag_lds(lds_h, 128, 96, lane);
        float skip[8];
#pragma unroll
        for (int r = 0; r < 8; ++r)
            skip[r] = (float)lds_h[(r + 8 * hi) * 128 + n0 + N];
        __syncthreads();   // all reads done before overwrites

        const float* bias = (layer == 0) ? bs0 : (layer == 1) ? bs1
                          : (layer == 2) ? bs2 : bs3;
        float bv = bias[n0 + N];
#pragma unroll
        for (int r = 0; r < 8; ++r) acc[r] = bv;

        const _Float16* Wf = wsh + OFF_WS + layer * 16384;
        acc = wmma_f16(a0, load_b_frag_pre(Wf, 4, 0, w, lane), acc);
        acc = wmma_f16(a1, load_b_frag_pre(Wf, 4, 1, w, lane), acc);
        acc = wmma_f16(a2, load_b_frag_pre(Wf, 4, 2, w, lane), acc);
        acc = wmma_f16(a3, load_b_frag_pre(Wf, 4, 3, w, lane), acc);

#pragma unroll
        for (int r = 0; r < 8; ++r) {
            int m = r + 8 * hi;
            float out = silu(acc[r]) + skip[r];
            if (layer == 3)
                H16[((long)(r0 + m)) * 128 + n0 + N] = (_Float16)out;
            else
                lds_h[m * 128 + n0 + N] = (_Float16)out;
        }
    }
}

// =========================================================================
// Kernel 5: fused head (128 -> 960) + 4-block rational spline + pdf.
// =========================================================================
__global__ void __launch_bounds__(256) out_spline_kernel(
        const _Float16* __restrict__ wsh,
        const float* __restrict__ b_out,
        const float* __restrict__ f,
        float* __restrict__ out) {
    __shared__ _Float16 lds_h[16 * 128];
    __shared__ float    lds_y[16 * 164];   // 160 cols + pad

    int tid  = threadIdx.x;
    int lane = tid & 31;
    int w    = tid >> 5;
    int r0   = blockIdx.x * 16;
    int N    = lane & 15;
    int hi   = lane >> 4;

    const _Float16* H16  = wsh + OFF_H;
    const _Float16* Wout = wsh + OFF_WOUT;

    // async stage h tile [16 x 128] f16 (4 KB): 16 bytes per thread
    async_load_b128(lds_offset(lds_h) + tid * 16, (unsigned)(tid * 16),
                    (unsigned long long)(H16 + (long)r0 * 128));
    wait_async0();
    __syncthreads();

    // A fragments reused across all 60 N-tiles
    v16h a0 = load_a_frag_lds(lds_h, 128,  0, lane);
    v16h a1 = load_a_frag_lds(lds_h, 128, 32, lane);
    v16h a2 = load_a_frag_lds(lds_h, 128, 64, lane);
    v16h a3 = load_a_frag_lds(lds_h, 128, 96, lane);

    for (int c = 0; c < 6; ++c) {
        if (c < 5)  // warm next chunk of fragment-ordered W_out
            __builtin_prefetch(Wout + (long)(c + 1) * 10 * 4 * 512, 0, 1);

        // ---- GEMM phase: Y[:, 160c : 160c+160] -------------------------
        for (int t = w; t < 10; t += 8) {
            int nt   = c * 10 + t;
            int col0 = nt * 16;
            v8f acc;
            float bv = b_out[col0 + N];
#pragma unroll
            for (int r = 0; r < 8; ++r) acc[r] = bv;
            acc = wmma_f16(a0, load_b_frag_pre(Wout, 4, 0, nt, lane), acc);
            acc = wmma_f16(a1, load_b_frag_pre(Wout, 4, 1, nt, lane), acc);
            acc = wmma_f16(a2, load_b_frag_pre(Wout, 4, 2, nt, lane), acc);
            acc = wmma_f16(a3, load_b_frag_pre(Wout, 4, 3, nt, lane), acc);
#pragma unroll
            for (int r = 0; r < 8; ++r)
                lds_y[(r + 8 * hi) * 164 + t * 16 + N] = acc[r];
        }
        __syncthreads();

        // ---- spline phase: 16 rows x 4 leads = 64 problems -------------
        if (tid < 64) {
            int row  = tid & 15;
            int ll   = tid >> 4;
            int lead = c * 4 + ll;
            const float* Yr = lds_y + row * 164 + ll * 40;
            float ff = f[(long)(r0 + row) * LEAD + lead];
            float dt_prod = 1.f;
#pragma unroll
            for (int blk = 0; blk < 4; ++blk) {
                const float* pr = Yr + blk * 10;
                float t_[5], y_[5];
                t_[0] = pr[0];
                y_[0] = pr[5];
#pragma unroll
                for (int i = 1; i < 5; ++i) {
                    t_[i] = t_[i - 1] + 0.001f + softplusf(pr[i]);
                    y_[i] = y_[i - 1] + 0.001f + softplusf(pr[5 + i]);
                }
                float df[4];
#pragma unroll
                for (int i = 0; i < 4; ++i)
                    df[i] = fdivf(y_[i + 1] - y_[i], t_[i + 1] - t_[i]);
                float d_[5];
                d_[0] = df[0] * df[0] * fdivf(t_[2] - t_[0], y_[2] - y_[0]);
#pragma unroll
                for (int i = 1; i < 4; ++i)
                    d_[i] = df[i - 1] * df[i] *
                            fdivf(t_[i + 1] - t_[i - 1], y_[i + 1] - y_[i - 1]);
                d_[4] = df[3] * df[3] * fdivf(t_[4] - t_[2], y_[4] - y_[2]);

                int idx = 0;
#pragma unroll
                for (int k = 0; k < 5; ++k) idx += (ff > t_[k]) ? 1 : 0;
                bool eq0 = (idx == 0), eq1 = (idx == 5);
                int k0 = min(max(idx - 1, 0), 3);
                float t0 = gsel5(t_, k0), t1 = gsel5(t_, k0 + 1);
                float y0 = gsel5(y_, k0), y1 = gsel5(y_, k0 + 1);
                float d0 = gsel5(d_, k0), d1 = gsel5(d_, k0 + 1);

                float dt_ = t1 - t0;
                float dy_ = y1 - y0;
                float rdt = frcp(dt_);
                float s   = dy_ * rdt;
                float e   = (ff - t0) * rdt;
                float oe  = 1.f - e;
                float b   = d1 + d0 - 2.f * s;
                float n0_ = dy_ * (s * e * e + d0 * e * oe);
                float n1_ = s + b * e * oe;
                float p_mid = y0 + n0_ * frcp(n1_);
                float p_lo  = d0 * ff + (y0 - d0 * t0);
                float p_hi  = d1 * ff + (y1 - d1 * t1);
                float val = eq0 ? p_lo : (eq1 ? p_hi : p_mid);
                float dv0 = s * s * (d1 * e * e + 2.f * s * e * oe + d0 * oe * oe);
                float der = eq0 ? d0 : (eq1 ? d1 : dv0 * frcp(n1_ * n1_));
                dt_prod *= der;
                ff = val;
            }
            out[(long)(r0 + row) * LEAD + lead] =
                __expf(-0.5f * ff * ff) * dt_prod * INV_SQRT2PI;
        }
        __syncthreads();
    }
}

// =========================================================================
extern "C" void kernel_launch(void* const* d_in, const int* in_sizes, int n_in,
                              void* d_out, int out_size, void* d_ws, size_t ws_size,
                              hipStream_t stream) {
    const float* x     = (const float*)d_in[0];
    const float* p     = (const float*)d_in[1];
    const float* f     = (const float*)d_in[2];
    const float* W_in  = (const float*)d_in[3];
    const float* b_in  = (const float*)d_in[4];
    const float* Ws0   = (const float*)d_in[5];
    const float* bs0   = (const float*)d_in[6];
    const float* Ws1   = (const float*)d_in[7];
    const float* bs1   = (const float*)d_in[8];
    const float* Ws2   = (const float*)d_in[9];
    const float* bs2   = (const float*)d_in[10];
    const float* Ws3   = (const float*)d_in[11];
    const float* bs3   = (const float*)d_in[12];
    const float* W_out = (const float*)d_in[13];
    const float* b_out = (const float*)d_in[14];

    _Float16* wsh = (_Float16*)d_ws;            // needs ~24.4 MB of d_ws
    float*    out = (float*)d_out;

    prep_weights<<<48, 256, 0, stream>>>(W_in, Ws0, Ws1, Ws2, Ws3, W_out, wsh);
    stats_kernel<<<(BATCH * LEAD) / 8, 256, 0, stream>>>(x, wsh + OFF_A);
    pack_p<<<(BATCH * 16) / 256, 256, 0, stream>>>(p, wsh + OFF_A);
    mlp_kernel<<<BATCH / 16, 256, 0, stream>>>(wsh, b_in, bs0, bs1, bs2, bs3,
                                               wsh + OFF_H, wsh + OFF_A);
    out_spline_kernel<<<BATCH / 16, 256, 0, stream>>>(wsh, b_out, f, out);
}
